// HierarchicalLayer_88098369176148
// MI455X (gfx1250) — matrix operand order repaired and checked by reading, not compiled, not run on hardware
//
#include <hip/hip_runtime.h>

#define N_NODES 20000
#define N_EDGES 200000
#define HIDD 64
#define NH 4
#define NA 5
#define OUTC (HIDD * (NA + 1)) /* 384 */

typedef __attribute__((ext_vector_type(16))) __bf16 v16bf;
typedef __attribute__((ext_vector_type(8)))  float  v8f;

// ---------- scalar helpers ----------
__device__ __forceinline__ unsigned short f2bf(float f) {
  unsigned u = __builtin_bit_cast(unsigned, f);
  u += 0x7FFFu + ((u >> 16) & 1u); // round-to-nearest-even
  return (unsigned short)(u >> 16);
}

// ---------- WMMA helpers (CDNA5 v_wmma_f32_16x16x32_bf16) ----------
__device__ __forceinline__ v8f wmma_bf(v16bf a, v16bf b, v8f c) {
  return __builtin_amdgcn_wmma_f32_16x16x32_bf16(false, a, false, b, (short)0, c,
                                                 false, false);
}

// Fragment-swizzled storage: each lane's fragment contents are 32 contiguous
// bytes -> single vector load (2x b128).
__device__ __forceinline__ v16bf load_B(const v16bf* __restrict__ Wf, int NT,
                                        int kc, int t, int lane) {
  return Wf[(kc * NT + t) * 32 + lane];
}
__device__ __forceinline__ v16bf load_A(const unsigned short* T, int chunk, int lane) {
  return ((const v16bf*)T)[chunk * 32 + lane];
}
// position of element (row m, col/k) inside a swizzled A-tile
__device__ __forceinline__ int swz_idx(int row, int k) {
  int chunk = k >> 5, kk = k & 31;
  int lt = row + (((kk >> 3) & 1) << 4);
  int hf = (kk & 7) | ((kk >> 4) << 3);
  return (chunk * 32 + lt) * 16 + hf;
}

// bias + relu on 4 C fragments (16x64), store swizzled bf16 A-tile in LDS
__device__ __forceinline__ void store_hid(unsigned short* hid, const v8f* c,
                                          const float* __restrict__ bias, int lane) {
  int n = lane & 15, hi8 = ((lane >> 4) & 1) << 3;
#pragma unroll
  for (int t = 0; t < 4; ++t) {
    float bb = bias[t * 16 + n];
#pragma unroll
    for (int r = 0; r < 8; ++r) {
      float v = fmaxf(c[t][r] + bb, 0.f);
      hid[swz_idx(r + hi8, t * 16 + n)] = f2bf(v);
    }
  }
}

// ---------- weights: pad + convert + swizzle -> [A][KC][NT][32][16] ----------
__global__ void pad_weights_kernel(const float* __restrict__ src,
                                   unsigned short* __restrict__ dst,
                                   int K, int Nn, int KC, int NT, long total) {
  long tid = (long)blockIdx.x * blockDim.x + threadIdx.x;
  if (tid >= total) return;
  int j    = (int)(tid & 15);
  int lane = (int)((tid >> 4) & 31);
  long rest = tid >> 9;               // a*KC*NT + kc*NT + t
  int t  = (int)(rest % NT);
  long r2 = rest / NT;
  int kc = (int)(r2 % KC);
  int a  = (int)(r2 / KC);
  int k = kc * 32 + ((j & 8) << 1) + ((lane & 16) >> 1) + (j & 7);  // lane>=16 holds K rows +8..+15, +24..+31
  int n = t * 16 + (lane & 15);
  float v = (k < K && n < Nn) ? src[((long)a * K + k) * Nn + n] : 0.f;
  dst[tid] = f2bf(v);
}

__global__ void fill_u32(unsigned* p, unsigned v, long n) {
  long tid = (long)blockIdx.x * blockDim.x + threadIdx.x;
  if (tid < n) p[tid] = v;
}
__global__ void copy_h_kernel(const float* __restrict__ h, float* __restrict__ out) {
  int tid = blockIdx.x * blockDim.x + threadIdx.x; // N*64
  int nd = tid >> 6, k = tid & 63;
  out[(long)nd * OUTC + k] = h[(long)nd * HIDD + k];
}

// ---------- CSR build (once per call) ----------
__global__ void count_deg_kernel(const int* __restrict__ dst, unsigned* __restrict__ deg) {
  int e = blockIdx.x * blockDim.x + threadIdx.x;
  if (e < N_EDGES) atomicAdd(&deg[dst[e]], 1u);
}

// single block; 1000 threads x 20 rows = 20000 rows (exact)
__global__ __launch_bounds__(1024) void scan_kernel(const unsigned* __restrict__ deg,
                                                    unsigned* __restrict__ rowptr,
                                                    unsigned* __restrict__ cursor) {
  __shared__ unsigned part[1024];
  int t = threadIdx.x;
  unsigned s = 0;
  if (t < 1000)
    for (int i = 0; i < 20; ++i) s += deg[t * 20 + i];
  part[t] = s;
  __syncthreads();
  for (int off = 1; off < 1024; off <<= 1) { // Hillis-Steele inclusive scan
    unsigned v = (t >= off) ? part[t - off] : 0u;
    __syncthreads();
    part[t] += v;
    __syncthreads();
  }
  if (t < 1000) {
    unsigned run = (t == 0) ? 0u : part[t - 1];
    for (int i = 0; i < 20; ++i) {
      int n = t * 20 + i;
      rowptr[n] = run;
      cursor[n] = run;
      run += deg[n];
    }
  }
  if (t == 0) rowptr[N_NODES] = N_EDGES;
}

__global__ void csr_fill_kernel(const int* __restrict__ src, const int* __restrict__ dst,
                                unsigned* __restrict__ cursor,
                                int* __restrict__ csr_src, int* __restrict__ epos) {
  int e = blockIdx.x * blockDim.x + threadIdx.x;
  if (e >= N_EDGES) return;
  unsigned pos = atomicAdd(&cursor[dst[e]], 1u);
  csr_src[pos] = src[e];
  epos[e] = (int)pos;
}

// ---------- per-edge attention scores via WMMA (one layer) ----------
// wave = 16 edges; block = 4 waves = 64 edges; grid = E/64 = 3125 (exact).
// Scores are written into CSR slots (epos) -> contiguous per destination node.
__global__ __launch_bounds__(128) void edge_score_kernel(
    const int* __restrict__ src, const int* __restrict__ dst,
    const int* __restrict__ epos,
    const float* __restrict__ pca, const float* __restrict__ pimg,
    const v16bf* __restrict__ Wp1f, const v16bf* __restrict__ Wp2f,
    const v16bf* __restrict__ Wp3f,
    const v16bf* __restrict__ Wq1f, const v16bf* __restrict__ Wq2f,
    const v16bf* __restrict__ Wq3f,
    const float* __restrict__ bp1, const float* __restrict__ bp2,
    const float* __restrict__ bp3,
    const float* __restrict__ bq1, const float* __restrict__ bq2,
    const float* __restrict__ bq3,
    float* __restrict__ ep, float* __restrict__ eq) {
  // per wave (ushort units): pca 1 chunk (512), pimg 4 chunks (2048), hid 2 (1024)
  __shared__ v16bf smem_v[4 * 224];
  const int w = threadIdx.x >> 5, lane = threadIdx.x & 31;
  unsigned short* fpca = (unsigned short*)(smem_v + w * 224);
  unsigned short* fpi  = fpca + 512;
  unsigned short* hid  = fpi + 2048;
  const int ebase = (blockIdx.x * 4 + w) * 16;

  // ---- stage edge features (bf16, swizzled A layout) ----
  for (int i = lane; i < 512; i += 32) { // concat(pca[src],pca[dst]) padded 18->32
    int m = i >> 5, k = i & 31, e = ebase + m;
    float v = 0.f;
    if (k < 9)       v = pca[(long)src[e] * 9 + k];
    else if (k < 18) v = pca[(long)dst[e] * 9 + (k - 9)];
    fpca[swz_idx(m, k)] = f2bf(v);
  }
  for (int i = lane; i < 2048; i += 32) { // concat(pimg[src],pimg[dst]) -> 128
    int m = i >> 7, k = i & 127, e = ebase + m;
    float v = (k < 64) ? pimg[(long)src[e] * 64 + k]
                       : pimg[(long)dst[e] * 64 + (k - 64)];
    fpi[swz_idx(m, k)] = f2bf(v);
  }
  __syncthreads();

  // ========== P branch (pca) ==========
  { // GEMM1: [16,32] x [32,64]
    v8f c[4] = {};
    v16bf a = load_A(fpca, 0, lane);
#pragma unroll
    for (int t = 0; t < 4; ++t)
      c[t] = wmma_bf(a, load_B(Wp1f, 4, 0, t, lane), c[t]);
    store_hid(hid, c, bp1, lane);
  }
  __syncthreads();
  { // GEMM2: [16,64] x [64,64]
    v16bf a0 = load_A(hid, 0, lane);
    v16bf a1 = load_A(hid, 1, lane);
    __syncthreads();
    v8f c[4] = {};
#pragma unroll
    for (int t = 0; t < 4; ++t) {
      c[t] = wmma_bf(a0, load_B(Wp2f, 4, 0, t, lane), c[t]);
      c[t] = wmma_bf(a1, load_B(Wp2f, 4, 1, t, lane), c[t]);
    }
    store_hid(hid, c, bp2, lane);
  }
  __syncthreads();
  { // GEMM3: [16,64] x [64,16(pad of 4)] -> leaky_relu -> CSR-slot scores
    v16bf a0 = load_A(hid, 0, lane);
    v16bf a1 = load_A(hid, 1, lane);
    v8f c = {};
    c = wmma_bf(a0, load_B(Wp3f, 1, 0, 0, lane), c);
    c = wmma_bf(a1, load_B(Wp3f, 1, 1, 0, lane), c);
    int n = lane & 15, hi8 = ((lane >> 4) & 1) << 3;
    if (n < NH) {
      float bb = bp3[n];
#pragma unroll
      for (int r = 0; r < 8; ++r) {
        float v = c[r] + bb;
        v = v > 0.f ? v : 0.01f * v;
        int e = ebase + r + hi8;
        ep[(long)epos[e] * NH + n] = v;
      }
    }
  }
  __syncthreads();

  // ========== Q branch (pimg) ==========
  { // GEMM1: [16,128] x [128,64]
    v8f c[4] = {};
#pragma unroll
    for (int kc = 0; kc < 4; ++kc) {
      v16bf a = load_A(fpi, kc, lane);
#pragma unroll
      for (int t = 0; t < 4; ++t)
        c[t] = wmma_bf(a, load_B(Wq1f, 4, kc, t, lane), c[t]);
    }
    store_hid(hid, c, bq1, lane);
  }
  __syncthreads();
  { // GEMM2
    v16bf a0 = load_A(hid, 0, lane);
    v16bf a1 = load_A(hid, 1, lane);
    __syncthreads();
    v8f c[4] = {};
#pragma unroll
    for (int t = 0; t < 4; ++t) {
      c[t] = wmma_bf(a0, load_B(Wq2f, 4, 0, t, lane), c[t]);
      c[t] = wmma_bf(a1, load_B(Wq2f, 4, 1, t, lane), c[t]);
    }
    store_hid(hid, c, bq2, lane);
  }
  __syncthreads();
  { // GEMM3
    v16bf a0 = load_A(hid, 0, lane);
    v16bf a1 = load_A(hid, 1, lane);
    v8f c = {};
    c = wmma_bf(a0, load_B(Wq3f, 1, 0, 0, lane), c);
    c = wmma_bf(a1, load_B(Wq3f, 1, 1, 0, lane), c);
    int n = lane & 15, hi8 = ((lane >> 4) & 1) << 3;
    if (n < NH) {
      float bb = bq3[n];
#pragma unroll
      for (int r = 0; r < 8; ++r) {
        float v = c[r] + bb;
        v = v > 0.f ? v : 0.01f * v;
        int e = ebase + r + hi8;
        eq[(long)epos[e] * NH + n] = v;
      }
    }
  }
}

// ---------- edge softmax per (node, head): max / exp+sum / normalize ----------
// No atomics: each destination's scores are contiguous in CSR order.
__global__ void softmax_kernel(const unsigned* __restrict__ rowptr,
                               float* __restrict__ ep, float* __restrict__ eq) {
  long tid = (long)blockIdx.x * blockDim.x + threadIdx.x; // 2*N*NH
  int branch = tid >= (long)N_NODES * NH;
  long idx = branch ? tid - (long)N_NODES * NH : tid;
  int n = (int)(idx >> 2), hd = (int)(idx & 3);
  float* eb = branch ? eq : ep;
  unsigned i0 = rowptr[n], i1 = rowptr[n + 1];
  if (i0 == i1) return;
  float m = -3.402823466e38f;
  for (unsigned i = i0; i < i1; ++i) m = fmaxf(m, eb[(long)i * NH + hd]);
  float s = 0.f;
  for (unsigned i = i0; i < i1; ++i) {
    float v = __expf(eb[(long)i * NH + hd] - m);
    eb[(long)i * NH + hd] = v;
    s += v;
  }
  float inv = 1.f / s;
  for (unsigned i = i0; i < i1; ++i) eb[(long)i * NH + hd] *= inv;
}

// ---------- gather-aggregate: op/oq[n,h,:] = sum_e a[e,h] * h[src[e],:] ----------
// thread = (node, k); in-edges contiguous; h[src] gather coalesced across k.
__global__ void aggregate_kernel(const unsigned* __restrict__ rowptr,
                                 const int* __restrict__ csr_src,
                                 const float* __restrict__ hprev, // d_out + a*64
                                 const float* __restrict__ ap, const float* __restrict__ aq,
                                 float* __restrict__ op, float* __restrict__ oq) {
  int tid = blockIdx.x * blockDim.x + threadIdx.x; // N*64
  int n = tid >> 6, k = tid & 63;
  unsigned i0 = rowptr[n], i1 = rowptr[n + 1];
  float accp[NH] = {0.f, 0.f, 0.f, 0.f};
  float accq[NH] = {0.f, 0.f, 0.f, 0.f};
  for (unsigned i = i0; i < i1; ++i) {
    float hv = hprev[(long)csr_src[i] * OUTC + k];
#pragma unroll
    for (int hd = 0; hd < NH; ++hd) {
      accp[hd] += hv * ap[(long)i * NH + hd];
      accq[hd] += hv * aq[(long)i * NH + hd];
    }
  }
#pragma unroll
  for (int hd = 0; hd < NH; ++hd) {
    op[((long)n * NH + hd) * HIDD + k] = accp[hd];
    oq[((long)n * NH + hd) * HIDD + k] = accq[hd];
  }
}

// ---------- x = relu(concat(op,oq) @ Wl + bl) via WMMA ----------
// wave = 16 nodes; block = 2 waves; grid = N/32 = 625 (exact).
__global__ __launch_bounds__(64) void node_linear_kernel(
    const float* __restrict__ op, const float* __restrict__ oq,
    const v16bf* __restrict__ Wlf, const float* __restrict__ bl,
    float* __restrict__ out) { // d_out + (a+1)*64
  __shared__ v16bf xs_v[2 * 512]; // per wave: 16 chunks x [32][16] halves
  int w = threadIdx.x >> 5, lane = threadIdx.x & 31;
  unsigned short* X = (unsigned short*)(xs_v + w * 512);
  long nb = ((long)blockIdx.x * 2 + w) * 16;
  for (int i = lane; i < 16 * 512; i += 32) {
    int m = i >> 9, k = i & 511;
    long node = nb + m;
    float v = (k < 256) ? op[node * 256 + k] : oq[node * 256 + (k - 256)];
    X[swz_idx(m, k)] = f2bf(v);
  }
  __syncthreads();
  v8f c[4] = {};
#pragma unroll
  for (int kc = 0; kc < 16; ++kc) {
    v16bf a = load_A(X, kc, lane);
#pragma unroll
    for (int t = 0; t < 4; ++t)
      c[t] = wmma_bf(a, load_B(Wlf, 4, kc, t, lane), c[t]);
  }
  int n = lane & 15, hi8 = ((lane >> 4) & 1) << 3;
#pragma unroll
  for (int t = 0; t < 4; ++t) {
    float bb = bl[t * 16 + n];
#pragma unroll
    for (int r = 0; r < 8; ++r) {
      float v = fmaxf(c[t][r] + bb, 0.f);
      out[(nb + r + hi8) * OUTC + t * 16 + n] = v;
    }
  }
}

// ---------- host launcher ----------
extern "C" void kernel_launch(void* const* d_in, const int* in_sizes, int n_in,
                              void* d_out, int out_size, void* d_ws, size_t ws_size,
                              hipStream_t stream) {
  const float* h    = (const float*)d_in[0];
  const float* pca  = (const float*)d_in[1];
  const float* pimg = (const float*)d_in[2];
  const float* Wp1  = (const float*)d_in[3];  const float* bp1 = (const float*)d_in[4];
  const float* Wp2  = (const float*)d_in[5];  const float* bp2 = (const float*)d_in[6];
  const float* Wp3  = (const float*)d_in[7];  const float* bp3 = (const float*)d_in[8];
  const float* Wq1  = (const float*)d_in[9];  const float* bq1 = (const float*)d_in[10];
  const float* Wq2  = (const float*)d_in[11]; const float* bq2 = (const float*)d_in[12];
  const float* Wq3  = (const float*)d_in[13]; const float* bq3 = (const float*)d_in[14];
  const float* Wl   = (const float*)d_in[15]; const float* bl  = (const float*)d_in[16];
  const int* src = (const int*)d_in[17];
  const int* dst = (const int*)d_in[18];
  float* out = (float*)d_out;

  // workspace layout (bytes); ~50 MB total
  char* ws = (char*)d_ws;
  size_t off = 0;
  auto take = [&](size_t bytes) { char* p = ws + off; off = (off + bytes + 255) & ~(size_t)255; return p; };
  // swizzled weights: [A][KC][NT][32][16] halves
  unsigned short* wp1f = (unsigned short*)take((size_t)NA * 1 * 4 * 512 * 2);
  unsigned short* wp2f = (unsigned short*)take((size_t)NA * 2 * 4 * 512 * 2);
  unsigned short* wp3f = (unsigned short*)take((size_t)NA * 2 * 1 * 512 * 2);
  unsigned short* wq1f = (unsigned short*)take((size_t)NA * 4 * 4 * 512 * 2);
  unsigned short* wq2f = (unsigned short*)take((size_t)NA * 2 * 4 * 512 * 2);
  unsigned short* wq3f = (unsigned short*)take((size_t)NA * 2 * 1 * 512 * 2);
  unsigned short* wlf  = (unsigned short*)take((size_t)NA * 16 * 4 * 512 * 2);
  unsigned* deg     = (unsigned*)take((size_t)N_NODES * 4);
  unsigned* rowptr  = (unsigned*)take((size_t)(N_NODES + 1) * 4);
  unsigned* cursor  = (unsigned*)take((size_t)N_NODES * 4);
  int*      csr_src = (int*)take((size_t)N_EDGES * 4);
  int*      epos    = (int*)take((size_t)N_EDGES * 4);
  float*    ep = (float*)take((size_t)N_EDGES * NH * 4);
  float*    eq = (float*)take((size_t)N_EDGES * NH * 4);
  float*    op = (float*)take((size_t)N_NODES * NH * HIDD * 4);
  float*    oq = (float*)take((size_t)N_NODES * NH * HIDD * 4);

  auto pad = [&](const float* s, unsigned short* d, int K, int Nn, int KC, int NT) {
    long total = (long)NA * KC * NT * 512;
    pad_weights_kernel<<<(unsigned)((total + 255) / 256), 256, 0, stream>>>(s, d, K, Nn, KC, NT, total);
  };
  pad(Wp1, wp1f, 18, 64, 1, 4);
  pad(Wp2, wp2f, 64, 64, 2, 4);
  pad(Wp3, wp3f, 64, 4, 2, 1);
  pad(Wq1, wq1f, 128, 64, 4, 4);
  pad(Wq2, wq2f, 64, 64, 2, 4);
  pad(Wq3, wq3f, 64, 4, 2, 1);
  pad(Wl, wlf, 512, 64, 16, 4);

  // CSR build (once per call)
  fill_u32<<<(N_NODES + 255) / 256, 256, 0, stream>>>(deg, 0u, (long)N_NODES);
  count_deg_kernel<<<(N_EDGES + 255) / 256, 256, 0, stream>>>(dst, deg);
  scan_kernel<<<1, 1024, 0, stream>>>(deg, rowptr, cursor);
  csr_fill_kernel<<<(N_EDGES + 255) / 256, 256, 0, stream>>>(src, dst, cursor, csr_src, epos);

  copy_h_kernel<<<(N_NODES * HIDD) / 256, 256, 0, stream>>>(h, out);

  for (int a = 0; a < NA; ++a) {
    edge_score_kernel<<<N_EDGES / 64, 128, 0, stream>>>(
        src, dst, epos, pca, pimg,
        (const v16bf*)(wp1f + (size_t)a * 1 * 4 * 512),
        (const v16bf*)(wp2f + (size_t)a * 2 * 4 * 512),
        (const v16bf*)(wp3f + (size_t)a * 2 * 1 * 512),
        (const v16bf*)(wq1f + (size_t)a * 4 * 4 * 512),
        (const v16bf*)(wq2f + (size_t)a * 2 * 4 * 512),
        (const v16bf*)(wq3f + (size_t)a * 2 * 1 * 512),
        bp1 + a * 64, bp2 + a * 64, bp3 + a * 4,
        bq1 + a * 64, bq2 + a * 64, bq3 + a * 4,
        ep, eq);

    softmax_kernel<<<(2 * N_NODES * NH) / 256, 256, 0, stream>>>(rowptr, ep, eq);

    aggregate_kernel<<<(N_NODES * HIDD) / 256, 256, 0, stream>>>(
        rowptr, csr_src, out + a * HIDD, ep, eq, op, oq);

    node_linear_kernel<<<N_NODES / 32, 64, 0, stream>>>(
        op, oq, (const v16bf*)(wlf + (size_t)a * 16 * 4 * 512), bl + a * 64,
        out + (a + 1) * HIDD);
  }
}
